// infer_LSTM_Decoder_41686952575788
// MI455X (gfx1250) — matrix operand order repaired and checked by reading, not compiled, not run on hardware
//
#include <hip/hip_runtime.h>
#include <hip/hip_bf16.h>
#include <math.h>

typedef __bf16 bf16;
typedef __attribute__((ext_vector_type(16))) __bf16 v16bf;
typedef __attribute__((ext_vector_type(8)))  float  v8f;
typedef __attribute__((ext_vector_type(4)))  unsigned int v4u;
typedef __attribute__((ext_vector_type(8)))  int    v8i;
typedef __attribute__((ext_vector_type(4)))  int    v4i;

#define NB 64
#define LL 196
#define AA 512
#define HH 512
#define EE 512
#define VV 30000
#define TT 20
#define START_TOK 1

__device__ __forceinline__ float sigmoidf_(float x) { return 1.f / (1.f + __expf(-x)); }

// ---------------------------------------------------------------------------
// bf16 WMMA GEMM with TDM (Tensor Data Mover) staging of B into LDS.
//   C = act(A @ B + bias);  A: MxK bf16 row-major, B: KxN bf16 row-major.
// Workgroup = 128 threads (4 waves) -> 64(M) x 16(N) output block.
// Wave 0 issues ONE tensor_load_to_lds for the whole Kx16 B-strip
// (2D descriptor: stride N, tile 16 x K, data_size 2B), waits TENSORcnt,
// then all waves consume it from LDS while A streams from global (b128).
// Requires M%64==0, N%16==0, K%32==0, K<=65535 (all GEMMs here qualify).
// ---------------------------------------------------------------------------
__global__ __launch_bounds__(128) void gemm_bf16_wmma(
    const bf16* __restrict__ A, const bf16* __restrict__ B,
    const float* __restrict__ bias, float* __restrict__ C,
    bf16* __restrict__ Cbf, int M, int N, int K, int act)
{
    extern __shared__ char smem[];
    bf16* ldsB = (bf16*)smem;                 // [K][16] row-major

    const int nt  = N >> 4;
    int mblk = blockIdx.x / nt;
    int tn   = blockIdx.x % nt;
    int wave = threadIdx.x >> 5;
    int lane = threadIdx.x & 31;
    int hi   = lane >> 4;                     // 0: lanes 0-15, 1: lanes 16-31
    int li   = lane & 15;
    int m = (mblk * 4 + wave) * 16 + li;      // A row owned by this lane
    int n = (tn << 4) + li;                   // B/C column owned by this lane
    const bf16* Arow = A + (size_t)m * K;

    if (threadIdx.x < 32) {
        // ---- Tensor DMA descriptor (D#), ISA ch.8 ----
        unsigned long long ga = (unsigned long long)(B + (size_t)(tn << 4));
        unsigned lds_base = (unsigned)(size_t)ldsB;   // LDS byte offset
        v4u g0;
        g0[0] = 1u;                                   // count=1, user mode
        g0[1] = lds_base;                             // lds_addr
        g0[2] = (unsigned)ga;                         // global_addr[31:0]
        g0[3] = (unsigned)((ga >> 32) & 0x1FFFFFFu) | (2u << 30); // addr[56:32] | type=2
        v8i g1;
        g1[0] = (int)(1u << 16);                      // data_size=1 -> 2 bytes
        g1[1] = (int)(((unsigned)N & 0xFFFFu) << 16); // tensor_dim0 lo16
        g1[2] = (int)(((unsigned)N >> 16) | (((unsigned)K & 0xFFFFu) << 16)); // dim0 hi | dim1 lo
        g1[3] = (int)(((unsigned)K >> 16) | (16u << 16)); // dim1 hi | tile_dim0=16
        g1[4] = (int)((unsigned)K & 0xFFFFu);         // tile_dim1=K, tile_dim2=0
        g1[5] = N;                                    // tensor_dim0_stride lo32
        g1[6] = 0;                                    // stride hi16 | dim1_stride lo16
        g1[7] = 0;
        v4i z4 = {0, 0, 0, 0};
        v8i z8 = {0, 0, 0, 0, 0, 0, 0, 0};
        __builtin_amdgcn_tensor_load_to_lds(g0, g1, z4, z4, z8, 0);
        __builtin_amdgcn_s_wait_tensorcnt(0);
    }
    __syncthreads();

    v8f acc = {};
    for (int k0 = 0; k0 < K; k0 += 32) {
        v16bf av, bv;
        // A fragment (ISA 16-bit A 16x32 layout): contiguous 8+8 halves
        const bf16* ap = Arow + k0 + hi * 8;
#pragma unroll
        for (int i = 0; i < 8; ++i) av[i] = ap[i];
#pragma unroll
        for (int i = 0; i < 8; ++i) av[8 + i] = ap[16 + i];
        if (k0 + 64 < K)
            __builtin_prefetch((const void*)(ap + 64), 0, 0);
        // B fragment from LDS strip: elem e -> K = k0 + hi*8 + e (e<8), k0 + 16 + hi*8 + (e-8) (e>=8), column li
#pragma unroll
        for (int e = 0; e < 16; ++e) bv[e] = ldsB[(size_t)(k0 + hi * 8 + (e < 8 ? e : e + 8)) * 16 + li];
        acc = __builtin_amdgcn_wmma_f32_16x16x32_bf16(
            false, av, false, bv, (short)0, acc, false, false);
    }

    float bsv = bias ? bias[n] : 0.f;
#pragma unroll
    for (int r = 0; r < 8; ++r) {
        int mm = (mblk * 4 + wave) * 16 + r + hi * 8;  // C/D layout: VGPR r -> M=r (+8 hi)
        float v = acc[r] + bsv;
        if (act == 1) v = tanhf(v);
        size_t idx = (size_t)mm * N + n;
        if (C)   C[idx]   = v;
        if (Cbf) Cbf[idx] = (bf16)v;
    }
}

// ---------------------------------------------------------------------------
__global__ void cvt_f32_bf16(const float* __restrict__ src, bf16* __restrict__ dst,
                             size_t count)
{
    size_t i = (size_t)blockIdx.x * blockDim.x + threadIdx.x;
    size_t stride = (size_t)gridDim.x * blockDim.x;
    for (; i < count; i += stride) dst[i] = (bf16)src[i];
}

// mean over L, store bf16 (GEMM input for h0/c0)
__global__ __launch_bounds__(512) void mean_kernel(const float* __restrict__ hl,
                                                   bf16* __restrict__ mean_bf)
{
    int n = blockIdx.x, a = threadIdx.x;
    float s = 0.f;
    for (int l = 0; l < LL; ++l)
        s += hl[((size_t)n * LL + l) * AA + a];
    mean_bf[n * AA + a] = (bf16)(s * (1.0f / (float)LL));
}

__global__ void init_tok_kernel(int* tok)
{
    int i = threadIdx.x;
    if (i < NB) tok[i] = START_TOK;
}

// ---------------------------------------------------------------------------
// Per-step attention: e -> softmax -> z, beta gate, build x = [z*beta, emb, h]
// One block per batch element n.
// ---------------------------------------------------------------------------
__global__ __launch_bounds__(256) void attention_kernel(
    const float* __restrict__ hl, const float* __restrict__ hlWa,
    const float* __restrict__ q,  const float* __restrict__ va,
    const float* __restrict__ h,  const float* __restrict__ Wb,
    const float* __restrict__ bb, const float* __restrict__ emb,
    const int* __restrict__ tok,  const bf16* __restrict__ h_bf,
    bf16* __restrict__ x_bf)
{
    __shared__ float e_sh[LL];
    __shared__ float red[256];
    int n = blockIdx.x, tid = threadIdx.x;
    const float* qn = q + (size_t)n * HH;

    // e[l] = sum_h va[h] * tanh(hlWa[n,l,h] + q[n,h])
    for (int l = tid; l < LL; l += 256) {
        const float* row = hlWa + ((size_t)n * LL + l) * HH;
        float s = 0.f;
        for (int hh = 0; hh < HH; ++hh)
            s += va[hh] * tanhf(row[hh] + qn[hh]);
        e_sh[l] = s;
    }
    __syncthreads();

    // softmax max
    float mloc = -1e30f;
    for (int l = tid; l < LL; l += 256) mloc = fmaxf(mloc, e_sh[l]);
    red[tid] = mloc; __syncthreads();
    for (int s = 128; s > 0; s >>= 1) {
        if (tid < s) red[tid] = fmaxf(red[tid], red[tid + s]);
        __syncthreads();
    }
    float mx = red[0]; __syncthreads();

    // exp + sum
    float sloc = 0.f;
    for (int l = tid; l < LL; l += 256) {
        float v = __expf(e_sh[l] - mx);
        e_sh[l] = v;
        sloc += v;
    }
    red[tid] = sloc; __syncthreads();
    for (int s = 128; s > 0; s >>= 1) {
        if (tid < s) red[tid] += red[tid + s];
        __syncthreads();
    }
    float inv_denom = 1.f / red[0]; __syncthreads();

    // beta = sigmoid(h . Wb + bb)
    float bloc = 0.f;
    for (int hh = tid; hh < HH; hh += 256) bloc += h[(size_t)n * HH + hh] * Wb[hh];
    red[tid] = bloc; __syncthreads();
    for (int s = 128; s > 0; s >>= 1) {
        if (tid < s) red[tid] += red[tid + s];
        __syncthreads();
    }
    float beta = sigmoidf_(red[0] + bb[0]); __syncthreads();

    // z[a] = beta * sum_l alpha[l]*hl[n,l,a]; x = [z, emb[tok], h]
    int tk = tok[n];
    bf16* xn = x_bf + (size_t)n * (AA + EE + HH);
    for (int a = tid; a < AA; a += 256) {
        float z = 0.f;
        for (int l = 0; l < LL; ++l)
            z += e_sh[l] * hl[((size_t)n * LL + l) * AA + a];
        z *= inv_denom * beta;
        xn[a]            = (bf16)z;
        xn[AA + a]       = (bf16)emb[(size_t)tk * EE + a];
        xn[AA + EE + a]  = h_bf[(size_t)n * HH + a];
    }
}

// LSTM pointwise update (keras gate order i,f,g,o)
__global__ __launch_bounds__(256) void lstm_kernel(
    const float* __restrict__ gates, float* __restrict__ h,
    float* __restrict__ c, bf16* __restrict__ h_bf)
{
    int idx = blockIdx.x * 256 + threadIdx.x;   // NB*HH threads
    int n = idx >> 9, j = idx & 511;
    const float* g = gates + (size_t)n * (4 * HH);
    float ig = sigmoidf_(g[j]);
    float fg = sigmoidf_(g[HH + j]);
    float gg = tanhf(g[2 * HH + j]);
    float og = sigmoidf_(g[3 * HH + j]);
    float cn = fg * c[idx] + ig * gg;
    float hn = og * tanhf(cn);
    c[idx] = cn;
    h[idx] = hn;
    h_bf[idx] = (bf16)hn;
}

// argmax over V with first-max tie-breaking (matches jnp.argmax)
__global__ __launch_bounds__(256) void argmax_kernel(
    const float* __restrict__ logits, int* __restrict__ tok,
    int* __restrict__ out, int t)
{
    __shared__ float bv[256];
    __shared__ int   bi[256];
    int n = blockIdx.x, tid = threadIdx.x;
    float best = -1e30f;
    int bidx = 0x7fffffff;
    for (int v = tid; v < VV; v += 256) {
        float val = logits[(size_t)n * VV + v];
        if (val > best) { best = val; bidx = v; }   // strided asc: first occurrence kept
    }
    bv[tid] = best; bi[tid] = bidx; __syncthreads();
    for (int s = 128; s > 0; s >>= 1) {
        if (tid < s) {
            if (bv[tid + s] > bv[tid] ||
                (bv[tid + s] == bv[tid] && bi[tid + s] < bi[tid])) {
                bv[tid] = bv[tid + s];
                bi[tid] = bi[tid + s];
            }
        }
        __syncthreads();
    }
    if (tid == 0) {
        tok[n] = bi[0];
        out[n * TT + t] = bi[0];
    }
}

// ---------------------------------------------------------------------------
extern "C" void kernel_launch(void* const* d_in, const int* in_sizes, int n_in,
                              void* d_out, int out_size, void* d_ws, size_t ws_size,
                              hipStream_t stream)
{
    const float* hl   = (const float*)d_in[0];   // [N,L,A]
    const float* emb  = (const float*)d_in[1];   // [V,E]
    const float* Wh0  = (const float*)d_in[2];   // [A,H]
    const float* bh0  = (const float*)d_in[3];   // [H]
    const float* Wc0  = (const float*)d_in[4];   // [A,H]
    const float* bc0  = (const float*)d_in[5];   // [H]
    const float* Wa   = (const float*)d_in[6];   // [A,H]
    const float* Ua   = (const float*)d_in[7];   // [H,H]
    const float* ba   = (const float*)d_in[8];   // [H]
    const float* va   = (const float*)d_in[9];   // [H]
    const float* Wb   = (const float*)d_in[10];  // [H,1]
    const float* bb   = (const float*)d_in[11];  // [1]
    const float* Wx   = (const float*)d_in[12];  // [A+E, 4H]
    const float* Wh   = (const float*)d_in[13];  // [H, 4H]
    const float* bl   = (const float*)d_in[14];  // [4H]
    const float* Wout = (const float*)d_in[15];  // [H,V]
    const float* bout = (const float*)d_in[16];  // [V]
    int* out = (int*)d_out;                      // [N,T] int32

    char* ws = (char*)d_ws;
    size_t off = 0;
    auto alloc = [&](size_t bytes) -> void* {
        off = (off + 255) & ~(size_t)255;
        void* p = ws + off;
        off += bytes;
        return p;
    };

    // bf16 weight copies (one-time; Wout_bf is 30.7 MB -> resident in 192 MB L2)
    bf16* Wh0b  = (bf16*)alloc((size_t)AA * HH * 2);
    bf16* Wc0b  = (bf16*)alloc((size_t)AA * HH * 2);
    bf16* Wab   = (bf16*)alloc((size_t)AA * HH * 2);
    bf16* Uab   = (bf16*)alloc((size_t)HH * HH * 2);
    bf16* Wxhb  = (bf16*)alloc((size_t)(AA + EE + HH) * 4 * HH * 2); // [Wx;Wh] stacked
    bf16* Woutb = (bf16*)alloc((size_t)HH * VV * 2);
    bf16* hlb   = (bf16*)alloc((size_t)NB * LL * AA * 2);
    // activations / state
    float* hlWa  = (float*)alloc((size_t)NB * LL * HH * 4);
    bf16* mean_b = (bf16*)alloc((size_t)NB * AA * 2);
    float* hbuf  = (float*)alloc((size_t)NB * HH * 4);
    float* cbuf  = (float*)alloc((size_t)NB * HH * 4);
    bf16* h_bf   = (bf16*)alloc((size_t)NB * HH * 2);
    float* qbuf  = (float*)alloc((size_t)NB * HH * 4);
    bf16* x_bf   = (bf16*)alloc((size_t)NB * (AA + EE + HH) * 2);
    float* gates = (float*)alloc((size_t)NB * 4 * HH * 4);
    float* logit = (float*)alloc((size_t)NB * VV * 4);
    int*   tok   = (int*)alloc((size_t)NB * 4);

    auto cvt = [&](const float* s, bf16* d, size_t cnt) {
        cvt_f32_bf16<<<1024, 256, 0, stream>>>(s, d, cnt);
    };
    // grid: (M/64) m-blocks x (N/16) n-tiles; dynamic LDS = Kx16 bf16 strip
    auto gemm = [&](const bf16* A, const bf16* B, const float* bias,
                    float* C, bf16* Cbf, int M, int N, int K, int act) {
        int blocks = (M / 64) * (N / 16);
        size_t shmem = (size_t)K * 16 * sizeof(bf16);
        gemm_bf16_wmma<<<blocks, 128, shmem, stream>>>(A, B, bias, C, Cbf, M, N, K, act);
    };

    // --- one-time setup ---
    cvt(Wh0, Wh0b, (size_t)AA * HH);
    cvt(Wc0, Wc0b, (size_t)AA * HH);
    cvt(Wa,  Wab,  (size_t)AA * HH);
    cvt(Ua,  Uab,  (size_t)HH * HH);
    cvt(Wx,  Wxhb,                              (size_t)(AA + EE) * 4 * HH);
    cvt(Wh,  Wxhb + (size_t)(AA + EE) * 4 * HH, (size_t)HH * 4 * HH);
    cvt(Wout, Woutb, (size_t)HH * VV);
    cvt(hl,   hlb,   (size_t)NB * LL * AA);

    mean_kernel<<<NB, 512, 0, stream>>>(hl, mean_b);
    // h0 = tanh(mean@Wh0+bh0), c0 = tanh(mean@Wc0+bc0)
    gemm(mean_b, Wh0b, bh0, hbuf, h_bf, NB, HH, AA, /*tanh*/1);
    gemm(mean_b, Wc0b, bc0, cbuf, nullptr, NB, HH, AA, /*tanh*/1);
    // attention keys: hlWa = hl @ Wa   [12544 x 512]
    gemm(hlb, Wab, nullptr, hlWa, nullptr, NB * LL, HH, AA, 0);
    init_tok_kernel<<<1, 64, 0, stream>>>(tok);

    // --- autoregressive decode ---
    for (int t = 0; t < TT; ++t) {
        // q = h @ Ua + ba
        gemm(h_bf, Uab, ba, qbuf, nullptr, NB, HH, HH, 0);
        // attention + beta gate + build x = [z*beta, emb[tok], h]
        attention_kernel<<<NB, 256, 0, stream>>>(hl, hlWa, qbuf, va, hbuf, Wb, bb,
                                                 emb, tok, h_bf, x_bf);
        // gates = x @ [Wx;Wh] + b_lstm
        gemm(x_bf, Wxhb, bl, gates, nullptr, NB, 4 * HH, AA + EE + HH, 0);
        lstm_kernel<<<(NB * HH) / 256, 256, 0, stream>>>(gates, hbuf, cbuf, h_bf);
        // logits = h @ Wout + bout  (heavy GEMM, bf16 WMMA, Wout strips via TDM)
        gemm(h_bf, Woutb, bout, logit, nullptr, NB, VV, HH, 0);
        argmax_kernel<<<NB, 256, 0, stream>>>(logit, tok, out, t);
    }
}
